// SNN_39874476376245
// MI455X (gfx1250) — matrix operand order repaired and checked by reading, not compiled, not run on hardware
//
#include <hip/hip_runtime.h>
#include <hip/hip_bf16.h>

// ---- problem constants (match reference) ----
#define B_    128
#define T_    16
#define IN_   2048
#define HID_  2048
#define OUT_  1024
#define DECAY_  0.2f
#define THRESH_ 0.5f
#define KT    64            // K-tile staged per LDS buffer

typedef __attribute__((ext_vector_type(16))) __bf16 v16bf;
typedef __attribute__((ext_vector_type(8)))  __bf16 v8bf;
typedef __attribute__((ext_vector_type(8)))  float  v8f;

// ---- bf16 helpers (bit-exact RNE convert; {0,1} spikes are exact) ----
static __device__ __forceinline__ __bf16 f2bf(float f) {
    unsigned u = __float_as_uint(f);
    u += 0x7fffu + ((u >> 16) & 1u);
    unsigned short h = (unsigned short)(u >> 16);
    return __builtin_bit_cast(__bf16, h);
}
static __device__ __forceinline__ float bf2f(__bf16 b) {
    unsigned short h = __builtin_bit_cast(unsigned short, b);
    return __uint_as_float(((unsigned)h) << 16);
}

// ---- CDNA5 async global->LDS (ASYNCcnt-tracked DMA, ISA ch.10 / 15.18) ----
// LDS operand is the wave-relative LDS byte offset; the generic pointer to a
// __shared__ object carries that offset in addr[31:0] (aperture tag is in the
// upper DWORD, ISA section 10.2), so truncation yields the LDS address.
static __device__ __forceinline__ unsigned lds_addr(const void* p) {
    return (unsigned)(uintptr_t)p;
}
static __device__ __forceinline__ void async_load_b128(unsigned lds, const void* gptr) {
    asm volatile("global_load_async_to_lds_b128 %0, %1, off"
                 :: "v"(lds), "v"(gptr) : "memory");
}
static __device__ __forceinline__ void wait_async(int pending) {
    if (pending) asm volatile("s_wait_asynccnt 0x6" ::: "memory");
    else         asm volatile("s_wait_asynccnt 0x0" ::: "memory");
}

// ---- LDS fragment loads (v_wmma_f32_16x16x32_bf16 operand layouts) ----
// A frag (16x32): lane m<16 -> row m, K {0..15}; lane m+16 -> {16..31}
static __device__ __forceinline__ v16bf lds_a_frag(const __bf16* Ab, int wave, int lane, int kk) {
    const int r  = wave * 16 + (lane & 15);
    const int kc = kk + ((lane >> 4) << 4);
    const __bf16* pa = Ab + r * KT + kc;
    union { v16bf v; v8bf h[2]; } u;
    u.h[0] = *(const v8bf*)(pa);
    u.h[1] = *(const v8bf*)(pa + 8);
    return u.v;
}
// B frag (32x16): lane n holds column n = 16 contiguous K values
static __device__ __forceinline__ v16bf lds_b_frag(const __bf16* Bb, int lane, int kk, int j) {
    const int c  = j * 16 + (lane & 15);
    const int kg = kk + ((lane >> 4) << 4);
    return *(const v16bf*)(Bb + c * KT + kg);
}

// =====================================================================
// Shared 128x64 output-tile GEMM core: C[128,64] += A[128,K] * W[64,K]^T
// 256 threads = 8 waves; wave w owns rows [16w,16w+16) x 4 16-wide N tiles.
// Double-buffered LDS staging via async DMA; WMMA bf16 16x16x32, f32 accum.
// Per staged tile: ALL ds_loads issued first, then 8 back-to-back WMMAs,
// so LDS latency is covered instead of a s_wait_dscnt 0 per WMMA.
// =====================================================================
static __device__ __forceinline__ void gemm_tile_128x64(
    const __bf16* __restrict__ A,   // [Mtot, ldk]
    const __bf16* __restrict__ W,   // [Ntot, ldk]
    int rowBlk, int col0, int ldk, int Ktot,
    __bf16* As,                     // __shared__ [2][128][KT]
    __bf16* Bs,                     // __shared__ [2][64][KT]
    v8f acc[4])
{
    const int tid  = threadIdx.x;
    const int lane = tid & 31;
    const int wave = tid >> 5;

    // issue one K-tile stage: A = 1024 16B chunks (4/thread), B = 512 (2/thread)
    auto stage = [&](int buf, int k0) {
#pragma unroll
        for (int s = 0; s < 4; ++s) {
            const int c = tid + s * 256;
            const int row = c >> 3, q = c & 7;          // 8 chunks per 64-elem row
            const __bf16* g = A + (size_t)(rowBlk + row) * ldk + k0 + q * 8;
            async_load_b128(lds_addr(As + (buf * 128 + row) * KT + q * 8), g);
        }
#pragma unroll
        for (int s = 0; s < 2; ++s) {
            const int c = tid + s * 256;
            const int row = c >> 3, q = c & 7;
            const __bf16* g = W + (size_t)(col0 + row) * ldk + k0 + q * 8;
            async_load_b128(lds_addr(Bs + (buf * 64 + row) * KT + q * 8), g);
        }
    };

    stage(0, 0);
    const int NT = Ktot / KT;
    for (int i = 0; i < NT; ++i) {
        const int buf = i & 1;
        const int more = (i + 1 < NT);
        if (more) stage(buf ^ 1, (i + 1) * KT);   // prefetch next tile
        wait_async(more ? 6 : 0);                 // tile i complete (in-order)
        __syncthreads();

        const __bf16* Ab = As + buf * 128 * KT;
        const __bf16* Bb = Bs + buf * 64 * KT;

        // ---- phase 1: all fragment loads (20x ds_load_b128) ----
        v16bf af[2];
        v16bf bfr[2][4];
#pragma unroll
        for (int s = 0; s < 2; ++s) {
            af[s] = lds_a_frag(Ab, wave, lane, s * 32);
#pragma unroll
            for (int j = 0; j < 4; ++j)
                bfr[s][j] = lds_b_frag(Bb, lane, s * 32, j);
        }
        // ---- phase 2: 8 back-to-back WMMAs ----
#pragma unroll
        for (int s = 0; s < 2; ++s)
#pragma unroll
            for (int j = 0; j < 4; ++j)
                acc[j] = __builtin_amdgcn_wmma_f32_16x16x32_bf16(
                    false, af[s], false, bfr[s][j], (short)0, acc[j], false, false);

        __syncthreads();   // all waves done reading buf before it is re-staged
    }
}

// ---- fp32 -> bf16 conversions ----
__global__ __launch_bounds__(256) void snn_cvt(const float* __restrict__ s,
                                               __bf16* __restrict__ d, int n) {
    int i = blockIdx.x * 256 + threadIdx.x;
    if (i < n) d[i] = f2bf(s[i]);
}
// x: [B,T,IN] fp32 -> xbf: [T,B,IN] bf16 (t-major so GEMM1 A is plain row-major)
__global__ __launch_bounds__(256) void snn_cvt_x(const float* __restrict__ x,
                                                 __bf16* __restrict__ xbf) {
    int i = blockIdx.x * 256 + threadIdx.x;
    if (i >= B_ * T_ * IN_) return;
    int c  = i % IN_;
    int bt = i / IN_;
    int t  = bt % T_;
    int b  = bt / T_;
    xbf[((size_t)t * B_ + b) * IN_ + c] = f2bf(x[i]);
}

// ---- GEMM1: H1pre[t*B+b, h] = x_t[b,:] . W1[h,:] + b1[h]  (done ONCE, reused 16x) ----
__global__ __launch_bounds__(256) void snn_fc1_gemm(const __bf16* __restrict__ Axt,
                                                    const __bf16* __restrict__ W1,
                                                    const float*  __restrict__ b1,
                                                    float* __restrict__ H1pre) {
    __shared__ __align__(64) __bf16 As[2 * 128 * KT];
    __shared__ __align__(64) __bf16 Bs[2 * 64 * KT];
    const int lane = threadIdx.x & 31;
    const int wave = threadIdx.x >> 5;
    const int rowBlk = blockIdx.x * 128;
    const int col0   = blockIdx.y * 64;

    v8f acc[4] = {};
    gemm_tile_128x64(Axt, W1, rowBlk, col0, IN_, IN_, As, Bs, acc);

    // C layout: VGPR r -> M = r + 8*(lane>=16), N = lane&15
    const int mHalf = (lane >> 4) << 3;
    const int nl    = lane & 15;
#pragma unroll
    for (int j = 0; j < 4; ++j) {
        const int n = col0 + j * 16 + nl;
        const float bias = b1[n];
#pragma unroll
        for (int r = 0; r < 8; ++r) {
            const int m = rowBlk + wave * 16 + mHalf + r;
            H1pre[(size_t)m * HID_ + n] = acc[j][r] + bias;
        }
    }
}

// ---- step1: layer-1 membrane update + spike (elementwise over B*HID) ----
__global__ __launch_bounds__(256) void snn_step1(const float* __restrict__ H1pre,
                                                 float* __restrict__ h1m,
                                                 __bf16* __restrict__ h1s, int t) {
    int i = blockIdx.x * 256 + threadIdx.x;
    if (i >= B_ * HID_) return;
    float sprev = bf2f(h1s[i]);                      // exactly 0.0 or 1.0
    float m = h1m[i] * DECAY_ * (1.0f - sprev) + H1pre[(size_t)t * (B_ * HID_) + i];
    h1m[i] = m;
    h1s[i] = f2bf((m > THRESH_) ? 1.0f : 0.0f);
}

// ---- step2: fused GEMM2 (h1s @ W2^T) + layer-2 membrane update + spike + acc ----
__global__ __launch_bounds__(256) void snn_step2(const __bf16* __restrict__ h1s,
                                                 const __bf16* __restrict__ W2,
                                                 const float*  __restrict__ b2,
                                                 float* __restrict__ h2m,
                                                 float* __restrict__ h2s,
                                                 float* __restrict__ outAcc,
                                                 int accumulate) {
    __shared__ __align__(64) __bf16 As[2 * 128 * KT];
    __shared__ __align__(64) __bf16 Bs[2 * 64 * KT];
    const int lane = threadIdx.x & 31;
    const int wave = threadIdx.x >> 5;
    const int col0 = blockIdx.x * 64;

    v8f acc[4] = {};
    gemm_tile_128x64(h1s, W2, /*rowBlk=*/0, col0, HID_, HID_, As, Bs, acc);

    const int mHalf = (lane >> 4) << 3;
    const int nl    = lane & 15;
#pragma unroll
    for (int j = 0; j < 4; ++j) {
        const int n = col0 + j * 16 + nl;
        const float bias = b2[n];
#pragma unroll
        for (int r = 0; r < 8; ++r) {
            const int m = wave * 16 + mHalf + r;
            const size_t idx = (size_t)m * OUT_ + n;
            float sprev = h2s[idx];
            float mem = h2m[idx] * DECAY_ * (1.0f - sprev) + acc[j][r] + bias;
            float spike = (mem > THRESH_) ? 1.0f : 0.0f;
            h2m[idx] = mem;
            h2s[idx] = spike;
            if (accumulate) outAcc[idx] += spike;
        }
    }
}

extern "C" void kernel_launch(void* const* d_in, const int* in_sizes, int n_in,
                              void* d_out, int out_size, void* d_ws, size_t ws_size,
                              hipStream_t stream) {
    (void)in_sizes; (void)n_in; (void)ws_size;
    const float* x  = (const float*)d_in[0];  // [B,T,IN]
    const float* w1 = (const float*)d_in[1];  // [HID,IN]
    const float* b1 = (const float*)d_in[2];  // [HID]
    const float* w2 = (const float*)d_in[3];  // [OUT,HID]
    const float* b2 = (const float*)d_in[4];  // [OUT]
    // d_in[5] = time_window (compile-time T_ here)

    // ---- workspace carve-up (~39 MB total) ----
    char* ws = (char*)d_ws;
    size_t off = 0;
    auto carve = [&](size_t bytes) -> void* {
        void* p = ws + off;
        off = (off + bytes + 255) & ~(size_t)255;
        return p;
    };
    __bf16* xbf   = (__bf16*)carve((size_t)T_ * B_ * IN_ * 2);   // 8 MB
    __bf16* w1bf  = (__bf16*)carve((size_t)HID_ * IN_ * 2);      // 8 MB
    __bf16* w2bf  = (__bf16*)carve((size_t)OUT_ * HID_ * 2);     // 4 MB
    float*  H1pre = (float*) carve((size_t)T_ * B_ * HID_ * 4);  // 16 MB
    float*  h1m   = (float*) carve((size_t)B_ * HID_ * 4);
    __bf16* h1s   = (__bf16*)carve((size_t)B_ * HID_ * 2);
    float*  h2m   = (float*) carve((size_t)B_ * OUT_ * 4);
    float*  h2s   = (float*) carve((size_t)B_ * OUT_ * 4);

    // ---- zero state + output (graph-capturable memsets) ----
    hipMemsetAsync(h1m, 0, (size_t)B_ * HID_ * 4, stream);
    hipMemsetAsync(h1s, 0, (size_t)B_ * HID_ * 2, stream);
    hipMemsetAsync(h2m, 0, (size_t)B_ * OUT_ * 4, stream);
    hipMemsetAsync(h2s, 0, (size_t)B_ * OUT_ * 4, stream);
    hipMemsetAsync(d_out, 0, (size_t)out_size * sizeof(float), stream);

    // ---- one-time conversions to bf16 ----
    snn_cvt_x<<<(B_ * T_ * IN_ + 255) / 256, 256, 0, stream>>>(x, xbf);
    snn_cvt<<<(HID_ * IN_ + 255) / 256, 256, 0, stream>>>(w1, w1bf, HID_ * IN_);
    snn_cvt<<<(OUT_ * HID_ + 255) / 256, 256, 0, stream>>>(w2, w2bf, OUT_ * HID_);

    // ---- hoisted fc1 GEMM: computed once, reused for all 16 outer iterations ----
    dim3 g1(T_ * B_ / 128, HID_ / 64);  // (16, 32)
    snn_fc1_gemm<<<g1, 256, 0, stream>>>(xbf, w1bf, b1, H1pre);

    // ---- sequential recurrence: 16 outer x 16 inner steps ----
    for (int outer = 0; outer < T_; ++outer) {
        const int accumulate = (outer == T_ - 1) ? 1 : 0;  // acc resets each outer; only last returned
        for (int t = 0; t < T_; ++t) {
            snn_step1<<<(B_ * HID_) / 256, 256, 0, stream>>>(H1pre, h1m, h1s, t);
            snn_step2<<<OUT_ / 64, 256, 0, stream>>>(h1s, w2bf, b2, h2m, h2s,
                                                     (float*)d_out, accumulate);
        }
    }
}
